// CKGConv_19739669692958
// MI455X (gfx1250) — compile-verified, matrix operand order checked by reading, not run on hardware
//
#include <hip/hip_runtime.h>
#include <hip/hip_bf16.h>
#include <math.h>

// ---------------------------------------------------------------------------
// CKGConv fused implementation for gfx1250 (MI455X), wave32 + WMMA f16.
//
// Heavy path: edge MLP  [E,24] -> [E,128] -> (GELU,128x128)x2 + residual
//             -> 128x128 final proj, all via v_wmma_f32_16x16x32_f16.
// GELU is the branch-free tanh form via v_exp_f32 / v_rcp_f32 (f16-accurate).
// gemm_tile is software-pipelined: all 8 fragment loads issued before the
// 4 chained WMMAs so a single wave has 16 loads in flight (partial waits).
// Fused epilogue: msg = xh[src] * score, atomic scatter-add into wV (L2-res).
// ---------------------------------------------------------------------------

typedef _Float16 v8h  __attribute__((ext_vector_type(8)));
typedef _Float16 v16h __attribute__((ext_vector_type(16)));
typedef float    v8f  __attribute__((ext_vector_type(8)));

#define HID  128
#define OUTD 64
#define IND  64
#define PED  24

union Frag16 { v16h v; v8h h[2]; };

// Branch-free GELU (tanh form): x * sigmoid(2u), u = 0.79788456(x+0.044715x^3)
// sigmoid(2u) = rcp(1 + exp2(-2u*log2(e))).  2 TRANS + ~5 VALU, no branches.
__device__ __forceinline__ float gelu_fast(float x) {
  float u = 0.79788456080286536f * __builtin_fmaf(0.044715f * x, x * x, x);
  float e = __builtin_amdgcn_exp2f(-2.8853900817779268f * u); // exp(-2u)
  return x * __builtin_amdgcn_rcpf(1.0f + e);
}

// ---------------------------------------------------------------------------
// Repack a [K x 128] f32 weight into f16 WMMA B-fragment order:
// tile (kt,nt) of 32x16 stored as [32 lanes][16 halfwords], where lane n(<16)
// holds col nt*16+n, K = kt*32 + {0..7, 16..23}; lane n+16 holds the same col,
// K = kt*32 + {8..15, 24..31}.  Rows >= Ksrc are zero-padded (K 24->32).
// ---------------------------------------------------------------------------
__global__ void prep_weight_kernel(const float* __restrict__ W,
                                   _Float16* __restrict__ Wt,
                                   int Ksrc, int total) {
  int i = blockIdx.x * blockDim.x + threadIdx.x;
  if (i >= total) return;
  int slot = i & 15;
  int lane = (i >> 4) & 31;
  int nt   = (i >> 9) & 7;
  int kt   = i >> 12;
  int klocal = (slot < 8 ? slot : slot + 8) + ((lane >= 16) ? 8 : 0);
  int k   = kt * 32 + klocal;
  int col = nt * 16 + (lane & 15);
  float v = (k < Ksrc) ? W[k * HID + col] : 0.0f;
  Wt[i] = (_Float16)v;
}

// xh = x @ W_x + b_x  : [N,64] x [64,128]  (small, bandwidth bound)
__global__ void xh_kernel(const float* __restrict__ x,
                          const float* __restrict__ Wx,
                          const float* __restrict__ bx,
                          float* __restrict__ xh, int N) {
  int i = blockIdx.x * blockDim.x + threadIdx.x;
  if (i >= N * HID) return;
  int n = i >> 7, c = i & 127;
  const float* xr = x + n * IND;
  float s = bx[c];
#pragma unroll 8
  for (int k = 0; k < IND; ++k) s += xr[k] * Wx[k * HID + c];
  xh[i] = s;
}

// per-target degree counts
__global__ void deg_kernel(const int* __restrict__ tgt, float* __restrict__ deg, int E) {
  int e = blockIdx.x * blockDim.x + threadIdx.x;
  if (e < E) atomicAdd(&deg[tgt[e]], 1.0f);
}

// ---------------------------------------------------------------------------
// Software-pipelined GEMM tile: issue ALL fragment loads (4x A from LDS,
// 4x B from global) before the chained WMMAs, so the wave has 16 outstanding
// loads and the compiler can use partial-count waits instead of full drains.
//
// A-fragment (wave-private LDS rows, stride-128 f16, per ISA layout):
//   lane l<16 : row l,    K = k0+{0..7}, k0+{16..23}
//   lane l>=16: row l-16, K = k0+{8..15}, k0+{24..31}
// ---------------------------------------------------------------------------
template <int NKT>
__device__ __forceinline__ v8f gemm_tile(const _Float16* rowbase,
                                         const _Float16* __restrict__ Wt,
                                         int nt, int lane) {
  const int half = lane >> 4;
  Frag16 a[NKT];
  v16h   b[NKT];
#pragma unroll
  for (int kt = 0; kt < NKT; ++kt) {
    const _Float16* rb = rowbase + kt * 32;
    a[kt].h[0] = *(const v8h*)(rb + half * 8);
    a[kt].h[1] = *(const v8h*)(rb + 16 + half * 8);
    b[kt] = *(const v16h*)(Wt + ((kt * 8 + nt) * 32 + lane) * 16);
  }
  v8f c = {};
#pragma unroll
  for (int kt = 0; kt < NKT; ++kt)
    c = __builtin_amdgcn_wmma_f32_16x16x32_f16(false, a[kt].v, false, b[kt],
                                               (short)0, c, false, false);
  return c;
}

// ---------------------------------------------------------------------------
// Fused edge-MLP + gather-multiply + scatter-add.
// Block = 256 threads = 8 waves; each wave owns a 16-edge M-tile (128 edges
// per block). Every wave touches only its own 16 LDS rows -> no barriers.
// x0 (pre-GELU input-proj output) is kept in 64 VGPRs for the residual.
// ---------------------------------------------------------------------------
__global__ __launch_bounds__(256)
void edge_mlp_wmma_kernel(const float* __restrict__ pe_val,
                          const int* __restrict__ pe_index,   // [2][E]
                          const _Float16* __restrict__ WtIn,
                          const _Float16* __restrict__ Wt1,
                          const _Float16* __restrict__ Wt2,
                          const _Float16* __restrict__ WtF,
                          const float* __restrict__ bIn, const float* __restrict__ b1,
                          const float* __restrict__ b2,  const float* __restrict__ bF,
                          const float* __restrict__ xh,
                          float* __restrict__ wV,
                          int E) {
  __shared__ _Float16 smem[32768];        // 64 KB
  _Float16* Hbuf  = smem;                 // [128][128] f16 activations (ping)
  _Float16* H2buf = smem + 16384;         // [128][128] f16 activations (pong)
  _Float16* A0buf = smem + 16384;         // [128][32]  PE staging, aliases H2
                                          // (dead before H2 is first written)
  const int tid   = threadIdx.x;
  const int w     = tid >> 5;
  const int lane  = tid & 31;
  const int ln    = lane & 15;
  const int half  = lane >> 4;
  const int eBase = blockIdx.x * 128 + w * 16;   // this wave's first edge

  // ---- stage PE tile (K padded 24 -> 32 with zeros), wave-private rows ----
  for (int i = lane; i < 512; i += 32) {
    int el = i >> 5, k = i & 31;
    int e = eBase + el;
    float v = (k < PED && e < E) ? pe_val[e * PED + k] : 0.0f;
    A0buf[(w * 16 + el) * 32 + k] = (_Float16)v;
  }

  const _Float16* A0row = A0buf + (w * 16 + ln) * 32;
  const _Float16* Hrow  = Hbuf  + (w * 16 + ln) * 128;
  const _Float16* H2row = H2buf + (w * 16 + ln) * 128;

  // ---- GEMM1: x0 = pe @ W_in + b_in (K=32 single step); keep x0 in regs ----
  v8f x0[8];
  {
    Frag16 a;
    a.h[0] = *(const v8h*)(A0row + half * 8);
    a.h[1] = *(const v8h*)(A0row + 16 + half * 8);
#pragma unroll
    for (int nt = 0; nt < 8; ++nt) {
      v16h b = *(const v16h*)(WtIn + (nt * 32 + lane) * 16);
      v8f c = {};
      c = __builtin_amdgcn_wmma_f32_16x16x32_f16(false, a.v, false, b,
                                                 (short)0, c, false, false);
      float bias = bIn[nt * 16 + ln];
#pragma unroll
      for (int r = 0; r < 8; ++r) c[r] += bias;
      x0[nt] = c;
#pragma unroll
      for (int r = 0; r < 8; ++r)
        Hbuf[(w * 16 + r + 8 * half) * 128 + nt * 16 + ln] =
            (_Float16)gelu_fast(c[r]);
    }
  }

  // ---- GEMM2: h1 = gelu(x0) @ W1 + b1 ; write gelu(h1) to H2 ----
#pragma unroll
  for (int nt = 0; nt < 8; ++nt) {
    v8f c = gemm_tile<4>(Hrow, Wt1, nt, lane);
    float bias = b1[nt * 16 + ln];
#pragma unroll
    for (int r = 0; r < 8; ++r)
      H2buf[(w * 16 + r + 8 * half) * 128 + nt * 16 + ln] =
          (_Float16)gelu_fast(c[r] + bias);
  }

  // ---- GEMM3: h2 = g(h1) @ W2 + b2 ; residual x0 += h2 ; write f16(x0) -> H
#pragma unroll
  for (int nt = 0; nt < 8; ++nt) {
    v8f c = gemm_tile<4>(H2row, Wt2, nt, lane);
    float bias = b2[nt * 16 + ln];
#pragma unroll
    for (int r = 0; r < 8; ++r) x0[nt][r] += c[r] + bias;
#pragma unroll
    for (int r = 0; r < 8; ++r)
      Hbuf[(w * 16 + r + 8 * half) * 128 + nt * 16 + ln] = (_Float16)x0[nt][r];
  }

  // ---- preload src/tgt for this wave's 16 edges (rows of the C layout) ----
  int srcs[8], tgts[8]; bool valid[8];
#pragma unroll
  for (int r = 0; r < 8; ++r) {
    int e = eBase + r + 8 * half;
    valid[r] = (e < E);
    tgts[r]  = valid[r] ? pe_index[e]     : 0;
    srcs[r]  = valid[r] ? pe_index[E + e] : 0;
  }

  // ---- GEMM4: score = x0 @ W_fin + b_fin ; fused gather-mul + scatter-add --
#pragma unroll
  for (int nt = 0; nt < 8; ++nt) {
    v8f c = gemm_tile<4>(Hrow, WtF, nt, lane);
    float bias = bF[nt * 16 + ln];
    int col = nt * 16 + ln;
#pragma unroll
    for (int r = 0; r < 8; ++r) {
      if (valid[r]) {
        float s  = c[r] + bias;
        float xv = xh[srcs[r] * HID + col];
        atomicAdd(&wV[tgts[r] * HID + col], s * xv);
      }
    }
  }
}

// out = ((wV/max(deg,1)) + head_bias) @ W_out + b_out
__global__ void finalize_kernel(const float* __restrict__ wV,
                                const float* __restrict__ deg,
                                const float* __restrict__ hb,
                                const float* __restrict__ Wout,
                                const float* __restrict__ bout,
                                float* __restrict__ out, int N) {
  int i = blockIdx.x * blockDim.x + threadIdx.x;
  if (i >= N * OUTD) return;
  int n = i >> 6, c = i & 63;
  float inv = 1.0f / fmaxf(deg[n], 1.0f);
  const float* wr = wV + n * HID;
  float s = bout[c];
#pragma unroll 8
  for (int j = 0; j < HID; ++j) s += (wr[j] * inv + hb[j]) * Wout[j * OUTD + c];
  out[i] = s;
}

extern "C" void kernel_launch(void* const* d_in, const int* in_sizes, int n_in,
                              void* d_out, int out_size, void* d_ws, size_t ws_size,
                              hipStream_t stream) {
  const float* x      = (const float*)d_in[0];
  const int*   pe_idx = (const int*)  d_in[1];
  const float* pe_val = (const float*)d_in[2];
  const float* W_in   = (const float*)d_in[3];
  const float* b_in   = (const float*)d_in[4];
  const float* W1     = (const float*)d_in[5];
  const float* b1     = (const float*)d_in[6];
  const float* W2     = (const float*)d_in[7];
  const float* b2     = (const float*)d_in[8];
  const float* W_fin  = (const float*)d_in[9];
  const float* b_fin  = (const float*)d_in[10];
  const float* W_x    = (const float*)d_in[11];
  const float* b_x    = (const float*)d_in[12];
  const float* head_b = (const float*)d_in[13];
  const float* W_out  = (const float*)d_in[14];
  const float* b_out  = (const float*)d_in[15];

  const int N = in_sizes[0] / IND;
  const int E = in_sizes[1] / 2;

  // workspace layout (256B aligned bumps)
  char* ws = (char*)d_ws;
  size_t off = 0;
  auto alloc = [&](size_t bytes) -> void* {
    void* p = ws + off;
    off = (off + bytes + 255) & ~(size_t)255;
    return p;
  };
  const int totIn  = 1 * 8 * 32 * 16;   // 4096 f16  (K padded 24->32)
  const int tot128 = 4 * 8 * 32 * 16;   // 16384 f16
  _Float16* WtIn = (_Float16*)alloc(totIn  * sizeof(_Float16));
  _Float16* Wt1  = (_Float16*)alloc(tot128 * sizeof(_Float16));
  _Float16* Wt2  = (_Float16*)alloc(tot128 * sizeof(_Float16));
  _Float16* WtF  = (_Float16*)alloc(tot128 * sizeof(_Float16));
  float* xh  = (float*)alloc((size_t)N * HID * sizeof(float));
  float* wV  = (float*)alloc((size_t)N * HID * sizeof(float));
  float* deg = (float*)alloc((size_t)N * sizeof(float));

  // weight repack (cheap; rerun every call for determinism)
  prep_weight_kernel<<<(totIn  + 255) / 256, 256, 0, stream>>>(W_in,  WtIn, PED, totIn);
  prep_weight_kernel<<<(tot128 + 255) / 256, 256, 0, stream>>>(W1,    Wt1,  HID, tot128);
  prep_weight_kernel<<<(tot128 + 255) / 256, 256, 0, stream>>>(W2,    Wt2,  HID, tot128);
  prep_weight_kernel<<<(tot128 + 255) / 256, 256, 0, stream>>>(W_fin, WtF,  HID, tot128);

  // zero accumulators
  hipMemsetAsync(wV,  0, (size_t)N * HID * sizeof(float), stream);
  hipMemsetAsync(deg, 0, (size_t)N * sizeof(float), stream);

  // node projection + degree counts
  xh_kernel<<<(N * HID + 255) / 256, 256, 0, stream>>>(x, W_x, b_x, xh, N);
  deg_kernel<<<(E + 255) / 256, 256, 0, stream>>>(pe_idx, deg, E);

  // fused heavy kernel: 128 edges / block
  int nblocks = (E + 127) / 128;
  edge_mlp_wmma_kernel<<<nblocks, 256, 0, stream>>>(
      pe_val, pe_idx, WtIn, Wt1, Wt2, WtF,
      b_in, b1, b2, b_fin, xh, wV, E);

  // mean + head bias + out projection
  finalize_kernel<<<(N * OUTD + 255) / 256, 256, 0, stream>>>(
      wV, deg, head_b, W_out, b_out, (float*)d_out, N);
}
